// AutoregressiveFlowSampler_79852031967623
// MI455X (gfx1250) — matrix operand order repaired and checked by reading, not compiled, not run on hardware
//
#include <hip/hip_runtime.h>
#include <hip/hip_bf16.h>
#include <math.h>

// ---------------------------------------------------------------------------
// Autoregressive transformer decode for MI455X (gfx1250, wave32, WMMA).
// One workgroup (4 waves, 128 threads) owns a 16-row batch tile and runs the
// whole T=64 sequential decode (batch rows are independent -> no grid sync).
// All dense GEMMs use v_wmma_f32_16x16x32_f16 (M=16 batch tile), fp32 accum.
// Attention (per-row KV history) is VALU with explicit 128-bit K/V loads.
// KV cache + fp16 weights live in d_ws:
//   ws layout (halves): [wqkv 196608][wo 65536][w1 262144][w2 262144]
//                       [Kcache 4*1024*64*128][Vcache 4*1024*64*128]
//   => ~130 MB workspace required.
// ---------------------------------------------------------------------------

#define LNUM   4
#define HNUM   4
#define DMODEL 128
#define DFFN   512
#define TSEQ   64
#define BTOT   1024
#define DHEAD  32
#define TILE_M 16

typedef _Float16 h16;
typedef __attribute__((ext_vector_type(16))) _Float16 v16h;
typedef __attribute__((ext_vector_type(8)))  _Float16 v8h;
typedef __attribute__((ext_vector_type(4)))  _Float16 v4h;
typedef __attribute__((ext_vector_type(8)))  float    v8f;

union HPack { v16h v; v8h h[2]; };

__device__ __forceinline__ v8f wmma16(v16h a, v16h b, v8f c) {
  // D = A(16x32 f16) * B(32x16 f16) + C(16x16 f32)
  return __builtin_amdgcn_wmma_f32_16x16x32_f16(false, a, false, b, (short)0, c,
                                                false, false);
}

// A fragment: row-major 16xK f16 (LDS). ISA 16-bit A layout:
// lanes 0-15 (m=lane):   halves 0..7 = K k0+0..7,  halves 8..15 = K k0+16..23
// lanes 16-31 (m=lane-16):halves 0..7 = K k0+8..15, halves 8..15 = K k0+24..31
__device__ __forceinline__ v16h load_a(const h16* base, int ldk, int k0, int lane) {
  const int m  = lane & 15;
  const int bo = (lane < 16) ? 0 : 8;
  const h16* p = base + m * ldk + k0 + bo;
  HPack u;
  u.h[0] = *(const v8h*)(p);
  u.h[1] = *(const v8h*)(p + 16);
  return u.v;
}

// B fragment from f16 W[N][K] (out = X @ W^T, so B[k][n] = W[n][k]).
// ISA 16-bit B layout: lane n holds column n; lanes 0-15 -> K k0+0..7, k0+16..23,
// lanes 16-31 -> K k0+8..15, k0+24..31 (same order as load_a).
__device__ __forceinline__ v16h load_b(const h16* W, int ldk, int n0, int k0, int lane) {
  const int n  = n0 + (lane & 15);
  const int kk = k0 + ((lane < 16) ? 0 : 8);
  const h16* p = W + (size_t)n * ldk + kk;
  HPack u;
  u.h[0] = *(const v8h*)(p);
  u.h[1] = *(const v8h*)(p + 16);
  return u.v;
}

// fp32 weight -> fp16 workspace conversion (one-shot preamble kernel)
__global__ void __launch_bounds__(256)
convert_weights_kernel(const float* __restrict__ wqkv, const float* __restrict__ wo,
                       const float* __restrict__ w1,   const float* __restrict__ w2,
                       h16* __restrict__ out) {
  const int idx = blockIdx.x * 256 + threadIdx.x;
  const int NQ = LNUM * 3 * DMODEL * DMODEL;   // 196608
  const int NO = LNUM * DMODEL * DMODEL;       // 65536
  const int N1 = LNUM * DFFN * DMODEL;         // 262144
  const int N2 = LNUM * DMODEL * DFFN;         // 262144
  if (idx < NQ)                     out[idx] = (h16)wqkv[idx];
  else if (idx < NQ + NO)           out[idx] = (h16)wo[idx - NQ];
  else if (idx < NQ + NO + N1)      out[idx] = (h16)w1[idx - NQ - NO];
  else if (idx < NQ + NO + N1 + N2) out[idx] = (h16)w2[idx - NQ - NO - N1];
}

__device__ __forceinline__ void ln_row_f16(const float* x, const float* w,
                                           const float* b, h16* o) {
  float mu = 0.f;
  for (int d = 0; d < DMODEL; ++d) mu += x[d];
  mu *= (1.0f / DMODEL);
  float var = 0.f;
  for (int d = 0; d < DMODEL; ++d) { float t = x[d] - mu; var += t * t; }
  var *= (1.0f / DMODEL);
  const float rs = rsqrtf(var + 1e-5f);
  for (int d = 0; d < DMODEL; ++d) o[d] = (h16)((x[d] - mu) * rs * w[d] + b[d]);
}

__global__ void __launch_bounds__(128)
decode_kernel(const float* __restrict__ gumbel,
              const float* __restrict__ state_emb,
              const float* __restrict__ pos_emb,
              const float* __restrict__ ln1w, const float* __restrict__ ln1b,
              const float* __restrict__ bqkv,
              const float* __restrict__ bo,
              const float* __restrict__ ln2w, const float* __restrict__ ln2b,
              const float* __restrict__ b1,   const float* __restrict__ b2,
              const float* __restrict__ fnw,  const float* __restrict__ fnb,
              const float* __restrict__ headw,const float* __restrict__ headb,
              const int* __restrict__ nalpha, const int* __restrict__ nbeta,
              const h16* __restrict__ Wq, const h16* __restrict__ Wo,
              const h16* __restrict__ W1, const h16* __restrict__ W2,
              h16* __restrict__ Kch, h16* __restrict__ Vch,
              float* __restrict__ out) {
  __shared__ float xbuf[TILE_M][DMODEL];                 // residual (f32)
  __shared__ __align__(16) h16 afrag[TILE_M][DMODEL];    // WMMA A staging
  __shared__ __align__(16) h16 hbuf[TILE_M][DFFN];       // FFN hidden (f16)
  __shared__ float obuf[TILE_M][DFFN];                   // GEMM outputs (f32)
  __shared__ float scb[TILE_M][HNUM][TSEQ];              // attn scores
  __shared__ int   tok[TILE_M];
  __shared__ int   arem[TILE_M];
  __shared__ int   brem[TILE_M];
  __shared__ float lpacc[TILE_M];

  const int tid  = threadIdx.x;
  const int lane = tid & 31;
  const int wave = tid >> 5;
  const int b0   = blockIdx.x * TILE_M;                  // first batch row

  if (tid < TILE_M) {
    tok[tid]   = 4;                                      // BOS
    arem[tid]  = nalpha[0];
    brem[tid]  = nbeta[0];
    lpacc[tid] = 0.f;
  }
  __syncthreads();

  for (int i = 0; i < TSEQ; ++i) {
    // ---- embed: x = state_emb[tok] + pos_emb[i]  (float4 vectorized) ----
    for (int idx = tid; idx < TILE_M * (DMODEL / 4); idx += 128) {
      const int m = idx >> 5, d = (idx & 31) * 4;
      const float4 se = *(const float4*)(state_emb + tok[m] * DMODEL + d);
      const float4 pe = *(const float4*)(pos_emb + i * DMODEL + d);
      float4 r;
      r.x = se.x + pe.x; r.y = se.y + pe.y;
      r.z = se.z + pe.z; r.w = se.w + pe.w;
      *(float4*)(&xbuf[m][d]) = r;
    }
    __syncthreads();

    for (int l = 0; l < LNUM; ++l) {
      // keep upcoming weights warm in L2 (global_prefetch_b8)
      __builtin_prefetch(Wq + (size_t)l * 3 * DMODEL * DMODEL + lane * 64, 0, 1);

      // ---- LN1 -> afrag (f16) ----------------------------------------
      if (tid < TILE_M)
        ln_row_f16(xbuf[tid], ln1w + l * DMODEL, ln1b + l * DMODEL, afrag[tid]);
      __syncthreads();

      // ---- QKV GEMM: 16x384 = afrag(16x128) @ Wqkv^T -----------------
      {
        const h16* W = Wq + (size_t)l * 3 * DMODEL * DMODEL;
        const float* bb = bqkv + l * 3 * DMODEL;
        const v16h a0 = load_a(&afrag[0][0], DMODEL, 0,  lane);
        const v16h a1 = load_a(&afrag[0][0], DMODEL, 32, lane);
        const v16h a2 = load_a(&afrag[0][0], DMODEL, 64, lane);
        const v16h a3 = load_a(&afrag[0][0], DMODEL, 96, lane);
        for (int nt = wave; nt < 24; nt += 4) {
          const int N0 = nt << 4;
          v8f c = {};
          c = wmma16(a0, load_b(W, DMODEL, N0, 0,  lane), c);
          c = wmma16(a1, load_b(W, DMODEL, N0, 32, lane), c);
          c = wmma16(a2, load_b(W, DMODEL, N0, 64, lane), c);
          c = wmma16(a3, load_b(W, DMODEL, N0, 96, lane), c);
          const int n = lane & 15, mr = (lane < 16) ? 0 : 8;
          const float bia = bb[N0 + n];
#pragma unroll
          for (int r = 0; r < 8; ++r) obuf[mr + r][N0 + n] = c[r] + bia;
        }
      }
      __syncthreads();

      // ---- write K,V (position i) to fp16 cache, 8B packed stores ----
      for (int idx = tid; idx < TILE_M * (DMODEL / 4); idx += 128) {
        const int m = idx >> 5, c4 = (idx & 31) * 4;
        const size_t row = (((size_t)l * BTOT + b0 + m) * TSEQ + i) * DMODEL;
        v4h kq, vq;
#pragma unroll
        for (int j = 0; j < 4; ++j) {
          kq[j] = (h16)obuf[m][DMODEL + c4 + j];
          vq[j] = (h16)obuf[m][2 * DMODEL + c4 + j];
        }
        *(v4h*)(Kch + row + c4) = kq;
        *(v4h*)(Vch + row + c4) = vq;
      }
      __threadfence();
      __syncthreads();

      // ---- attention: each of 64 threads owns one (row m, head h) ----
      if (tid < TILE_M * HNUM) {
        const int m = tid & 15, h = tid >> 4;
        const size_t rb = ((size_t)l * BTOT + b0 + m) * TSEQ;
        const h16* Kp = Kch + rb * DMODEL + h * DHEAD;
        const h16* Vp = Vch + rb * DMODEL + h * DHEAD;
        float qv[DHEAD];
#pragma unroll
        for (int d = 0; d < DHEAD; ++d) qv[d] = obuf[m][h * DHEAD + d];
        float mx = -1e30f;
        for (int t = 0; t <= i; ++t) {
          union { v8h v[4]; h16 e[DHEAD]; } kk;
          const v8h* kp = (const v8h*)(Kp + (size_t)t * DMODEL);
          kk.v[0] = kp[0]; kk.v[1] = kp[1]; kk.v[2] = kp[2]; kk.v[3] = kp[3];
          float s = 0.f;
#pragma unroll
          for (int d = 0; d < DHEAD; ++d) s += (float)kk.e[d] * qv[d];
          s *= 0.17677669529663687f;                     // 1/sqrt(32)
          scb[m][h][t] = s;
          mx = fmaxf(mx, s);
        }
        float sum = 0.f, cv[DHEAD];
#pragma unroll
        for (int d = 0; d < DHEAD; ++d) cv[d] = 0.f;
        for (int t = 0; t <= i; ++t) {
          const float w = __expf(scb[m][h][t] - mx);
          sum += w;
          union { v8h v[4]; h16 e[DHEAD]; } vv;
          const v8h* vp = (const v8h*)(Vp + (size_t)t * DMODEL);
          vv.v[0] = vp[0]; vv.v[1] = vp[1]; vv.v[2] = vp[2]; vv.v[3] = vp[3];
#pragma unroll
          for (int d = 0; d < DHEAD; ++d) cv[d] += w * (float)vv.e[d];
        }
        const float rs = 1.0f / sum;
#pragma unroll
        for (int d = 0; d < DHEAD; ++d)
          afrag[m][h * DHEAD + d] = (h16)(cv[d] * rs);
      }
      __syncthreads();

      // ---- WO GEMM: x += ctx @ Wo^T + bo -----------------------------
      {
        const h16* W = Wo + (size_t)l * DMODEL * DMODEL;
        const float* bb = bo + l * DMODEL;
        const v16h a0 = load_a(&afrag[0][0], DMODEL, 0,  lane);
        const v16h a1 = load_a(&afrag[0][0], DMODEL, 32, lane);
        const v16h a2 = load_a(&afrag[0][0], DMODEL, 64, lane);
        const v16h a3 = load_a(&afrag[0][0], DMODEL, 96, lane);
        for (int nt = wave; nt < 8; nt += 4) {
          const int N0 = nt << 4;
          v8f c = {};
          c = wmma16(a0, load_b(W, DMODEL, N0, 0,  lane), c);
          c = wmma16(a1, load_b(W, DMODEL, N0, 32, lane), c);
          c = wmma16(a2, load_b(W, DMODEL, N0, 64, lane), c);
          c = wmma16(a3, load_b(W, DMODEL, N0, 96, lane), c);
          const int n = lane & 15, mr = (lane < 16) ? 0 : 8;
          const float bia = bb[N0 + n];
#pragma unroll
          for (int r = 0; r < 8; ++r) xbuf[mr + r][N0 + n] += c[r] + bia;
        }
      }
      __syncthreads();

      // ---- LN2 -> afrag (f16) ----------------------------------------
      if (tid < TILE_M)
        ln_row_f16(xbuf[tid], ln2w + l * DMODEL, ln2b + l * DMODEL, afrag[tid]);
      __syncthreads();

      // ---- FFN W1 + exact GELU: hbuf(16x512 f16) ---------------------
      {
        const h16* W = W1 + (size_t)l * DFFN * DMODEL;
        const float* bb = b1 + l * DFFN;
        const v16h a0 = load_a(&afrag[0][0], DMODEL, 0,  lane);
        const v16h a1 = load_a(&afrag[0][0], DMODEL, 32, lane);
        const v16h a2 = load_a(&afrag[0][0], DMODEL, 64, lane);
        const v16h a3 = load_a(&afrag[0][0], DMODEL, 96, lane);
        for (int nt = wave; nt < 32; nt += 4) {
          const int N0 = nt << 4;
          v8f c = {};
          c = wmma16(a0, load_b(W, DMODEL, N0, 0,  lane), c);
          c = wmma16(a1, load_b(W, DMODEL, N0, 32, lane), c);
          c = wmma16(a2, load_b(W, DMODEL, N0, 64, lane), c);
          c = wmma16(a3, load_b(W, DMODEL, N0, 96, lane), c);
          const int n = lane & 15, mr = (lane < 16) ? 0 : 8;
          const float bia = bb[N0 + n];
#pragma unroll
          for (int r = 0; r < 8; ++r) {
            const float v = c[r] + bia;
            const float g = 0.5f * v * (1.0f + erff(v * 0.70710678118654752f));
            hbuf[mr + r][N0 + n] = (h16)g;
          }
        }
      }
      __syncthreads();

      // ---- FFN W2: x += h(16x512) @ W2^T + b2 ------------------------
      {
        const h16* W = W2 + (size_t)l * DMODEL * DFFN;
        const float* bb = b2 + l * DMODEL;
        for (int nt = wave; nt < 8; nt += 4) {
          const int N0 = nt << 4;
          v8f c = {};
#pragma unroll
          for (int kc = 0; kc < 16; ++kc) {
            const v16h a = load_a(&hbuf[0][0], DFFN, kc * 32, lane);
            c = wmma16(a, load_b(W, DFFN, N0, kc * 32, lane), c);
          }
          const int n = lane & 15, mr = (lane < 16) ? 0 : 8;
          const float bia = bb[N0 + n];
#pragma unroll
          for (int r = 0; r < 8; ++r) xbuf[mr + r][N0 + n] += c[r] + bia;
        }
      }
      __syncthreads();
    } // layers

    // ---- final LN + head + masked gumbel-max sampling -----------------
    if (tid < TILE_M) {
      const int m = tid;
      float mu = 0.f;
      for (int d = 0; d < DMODEL; ++d) mu += xbuf[m][d];
      mu *= (1.0f / DMODEL);
      float var = 0.f;
      for (int d = 0; d < DMODEL; ++d) { float t = xbuf[m][d] - mu; var += t * t; }
      var *= (1.0f / DMODEL);
      const float rs = rsqrtf(var + 1e-5f);
      for (int d = 0; d < DMODEL; ++d)
        obuf[m][d] = (xbuf[m][d] - mu) * rs * fnw[d] + fnb[d];

      float z[4];
      for (int s = 0; s < 4; ++s) {
        float acc = headb[s];
        for (int d = 0; d < DMODEL; ++d) acc += obuf[m][d] * headw[s * DMODEL + d];
        z[s] = acc;
      }
      const int oa = TSEQ - i - 1;
      const int a = arem[m], b = brem[m];
      bool mk[4];
      mk[0] = (a <= oa) && (b <= oa);
      mk[1] = (b > 0) && (a <= oa) && (b - 1 <= oa);
      mk[2] = (a > 0) && (a - 1 <= oa) && (b <= oa);
      mk[3] = (a > 0) && (b > 0) && (a - 1 <= oa) && (b - 1 <= oa);
      float mx = -1e30f;
      for (int s = 0; s < 4; ++s) if (mk[s] && z[s] > mx) mx = z[s];
      float se = 0.f;
      for (int s = 0; s < 4; ++s) if (mk[s]) se += __expf(z[s] - mx);
      const float lse = mx + __logf(se);
      const float* g = gumbel + ((size_t)(b0 + m) * TSEQ + i) * 4;
      float best = -1e30f, lpb = 0.f;
      int bs = 0;
      for (int s = 0; s < 4; ++s) {
        const float lp = mk[s] ? (z[s] - lse) : -1e30f;
        const float v  = lp + g[s];
        if (v > best) { best = v; bs = s; lpb = lp; }
      }
      lpacc[m] += lpb;
      tok[m]  = bs;
      arem[m] = a - (bs >> 1);
      brem[m] = b - (bs & 1);
      out[(size_t)(b0 + m) * TSEQ + i] = (float)bs;   // states (B,T)
    }
    __syncthreads();
  } // steps

  if (tid < TILE_M)
    out[(size_t)BTOT * TSEQ + b0 + tid] = lpacc[tid];  // log-probs (B,)
}

extern "C" void kernel_launch(void* const* d_in, const int* in_sizes, int n_in,
                              void* d_out, int out_size, void* d_ws, size_t ws_size,
                              hipStream_t stream) {
  const float* gumbel    = (const float*)d_in[0];
  const float* state_emb = (const float*)d_in[1];
  const float* pos_emb   = (const float*)d_in[2];
  const float* ln1w      = (const float*)d_in[3];
  const float* ln1b      = (const float*)d_in[4];
  const float* wqkv      = (const float*)d_in[5];
  const float* bqkv      = (const float*)d_in[6];
  const float* wo        = (const float*)d_in[7];
  const float* bo        = (const float*)d_in[8];
  const float* ln2w      = (const float*)d_in[9];
  const float* ln2b      = (const float*)d_in[10];
  const float* w1        = (const float*)d_in[11];
  const float* b1        = (const float*)d_in[12];
  const float* w2        = (const float*)d_in[13];
  const float* b2        = (const float*)d_in[14];
  const float* fnw       = (const float*)d_in[15];
  const float* fnb       = (const float*)d_in[16];
  const float* headw     = (const float*)d_in[17];
  const float* headb     = (const float*)d_in[18];
  const int*   nalpha    = (const int*)d_in[19];
  const int*   nbeta     = (const int*)d_in[20];

  h16* wsh = (h16*)d_ws;
  const size_t NQ = (size_t)LNUM * 3 * DMODEL * DMODEL;
  const size_t NO = (size_t)LNUM * DMODEL * DMODEL;
  const size_t N1 = (size_t)LNUM * DFFN * DMODEL;
  const size_t N2 = (size_t)LNUM * DMODEL * DFFN;
  h16* Wq  = wsh;
  h16* Wo  = Wq + NQ;
  h16* W1p = Wo + NO;
  h16* W2p = W1p + N1;
  h16* Kch = W2p + N2;
  h16* Vch = Kch + (size_t)LNUM * BTOT * TSEQ * DMODEL;
  // total ws need: (786432 + 2*33554432) halves ~= 130 MB

  convert_weights_kernel<<<3072, 256, 0, stream>>>(wqkv, wo, w1, w2, wsh);
  decode_kernel<<<BTOT / TILE_M, 128, 0, stream>>>(
      gumbel, state_emb, pos_emb, ln1w, ln1b, bqkv, bo, ln2w, ln2b, b1, b2,
      fnw, fnb, headw, headb, nalpha, nbeta,
      Wq, Wo, W1p, W2p, Kch, Vch, (float*)d_out);
}
